// AttentionSuper_66821101191193
// MI455X (gfx1250) — matrix operand order repaired and checked by reading, not compiled, not run on hardware
//
#include <hip/hip_runtime.h>

typedef __attribute__((ext_vector_type(16))) __bf16 v16bf;
typedef __attribute__((ext_vector_type(8)))  __bf16 v8bf;
typedef __attribute__((ext_vector_type(8)))  float  v8f;

#define DEVINL __device__ __forceinline__

constexpr int Bb = 64, Nn = 197, Cc = 640, Hh = 10, Dd = 64;
constexpr int Mm = Bb * Nn;        // 12608 rows (multiple of 16)
constexpr int NQ = 208;            // query dim padded to 13*16
constexpr int NK = 224;            // key dim padded to 7*32 (14*16)
constexpr int O3 = 3 * Cc;         // 1920
constexpr int GRID_S = 14;         // sqrt(196)
constexpr int MRP_ = 14;
constexpr float SCALE = 0.125f;    // D^-0.5 = 1/8
constexpr int BH = Bb * Hh;        // 640

// ---------------- WMMA helpers (CDNA5 bf16 16x16x32) ----------------

DEVINL v8f wmma_bf16(v16bf a, v16bf b, v8f c) {
  // 8 args: (neg_a, A, neg_b, B, c_mod, C, reuse_a, reuse_b)
  return __builtin_amdgcn_wmma_f32_16x16x32_bf16(false, a, false, b,
                                                 (short)0, c, false, false);
}

// A operand: 16x32 tile from row-major [rows][ld] bf16.
// ISA layout: lanes 0-15 hold M=lane, K in {0..7, 16..23}; lanes 16-31: K in {8..15, 24..31}
DEVINL v16bf load_a_bf16(const __bf16* base, int ld, int row0, int k0, int lane) {
  const __bf16* p = base + (size_t)(row0 + (lane & 15)) * ld + k0 + ((lane >> 4) << 3);
  v8bf lo = *(const v8bf*)p;
  v8bf hi = *(const v8bf*)(p + 16);
  v16bf r;
#pragma unroll
  for (int i = 0; i < 8; ++i) { r[i] = lo[i]; r[i + 8] = hi[i]; }
  return r;
}

// A operand loaded from fp32 source with on-the-fly bf16 conversion
DEVINL v16bf load_a_f32(const float* base, int ld, int row0, int k0, int lane) {
  const float* p = base + (size_t)(row0 + (lane & 15)) * ld + k0 + ((lane >> 4) << 3);
  v16bf r;
#pragma unroll
  for (int i = 0; i < 8; ++i) { r[i] = (__bf16)p[i]; r[i + 8] = (__bf16)p[i + 16]; }
  return r;
}

// B operand: B[k][n] with source row-major [n][k] (two 16B chunks per lane)
// ISA layout: lane%16 = N column, lanes 0-15 K in {0..7, 16..23}; lanes 16-31: K in {8..15, 24..31}
DEVINL v16bf load_b_nt(const __bf16* base, int ld, int n0, int k0, int lane) {
  const __bf16* p = base + (size_t)(n0 + (lane & 15)) * ld + k0 + ((lane >> 4) << 3);
  v8bf lo = *(const v8bf*)p;
  v8bf hi = *(const v8bf*)(p + 16);
  v16bf r;
#pragma unroll
  for (int i = 0; i < 8; ++i) { r[i] = lo[i]; r[i + 8] = hi[i]; }
  return r;
}

// ---------------- K0: weight conversion + qkv buffer zero ----------------

__global__ void prep_kernel(const float* __restrict__ wqkv, const float* __restrict__ wproj,
                            __bf16* wqkvb, __bf16* wprojb, __bf16* qkvb) {
  int tid = blockIdx.x * blockDim.x + threadIdx.x;
  int np  = gridDim.x * blockDim.x;
  for (int i = tid; i < O3 * Cc; i += np) wqkvb[i] = (__bf16)wqkv[i];
  for (int i = tid; i < Cc * Cc; i += np) wprojb[i] = (__bf16)wproj[i];
  for (size_t i = tid; i < (size_t)3 * BH * NK * Dd; i += np) qkvb[i] = (__bf16)0.f;
}

// ---------------- K1: rel-pos embedding expansion (AutoFormer 2D) ----------------
// rpkb is written key-row-major [i][k][d] (NT operand for the bias GEMM).
// rpvb is written TRANSPOSED [i][d][k] so the P@rpv GEMM B-operand is contiguous.

__global__ void relemb_kernel(const float* __restrict__ rkv, const float* __restrict__ rkh,
                              const float* __restrict__ rvv, const float* __restrict__ rvh,
                              __bf16* rpkb, __bf16* rpvt) {
  int idx = blockIdx.x * blockDim.x + threadIdx.x;
  if (idx >= Nn * NK) return;
  int i = idx / NK, kk = idx % NK;
  __bf16* pk = rpkb + (size_t)idx * Dd;                    // [i][kk][d]
  __bf16* pv = rpvt + ((size_t)i * Dd) * NK + kk;          // [i][d][kk], stride NK over d
  if (kk >= Nn) {
#pragma unroll 8
    for (int d = 0; d < Dd; ++d) { pk[d] = (__bf16)0.f; pv[(size_t)d * NK] = (__bf16)0.f; }
    return;
  }
  int fv = 0, fh = 0;  // index 0 = cls row/col
  if (i > 0 && kk > 0) {
    int iq = i - 1, ik = kk - 1;
    int dv = ik / GRID_S - iq / GRID_S;
    int dh = ik % GRID_S - iq % GRID_S;
    dv = dv < -MRP_ ? -MRP_ : (dv > MRP_ ? MRP_ : dv);
    dh = dh < -MRP_ ? -MRP_ : (dh > MRP_ ? MRP_ : dh);
    fv = dv + MRP_ + 1;
    fh = dh + MRP_ + 1;
  }
#pragma unroll 8
  for (int d = 0; d < Dd; ++d) {
    pk[d] = (__bf16)(rkv[fv * Dd + d] + rkh[fh * Dd + d]);
    pv[(size_t)d * NK] = (__bf16)(rvv[fv * Dd + d] + rvh[fh * Dd + d]);
  }
}

// ---------------- K2: QKV projection GEMM (WMMA) ----------------
// C[m][o] = sum_c x[m][c] * wqkv[o][c].
// Q,K scatter into [bh][n][d]; V scatters TRANSPOSED into [bh][d][n] so the
// P@V GEMM B-operand is contiguous. n is padded to NK (zero-filled by K0).

__global__ void __launch_bounds__(32) qkv_gemm(const float* __restrict__ x,
                                               const __bf16* __restrict__ wqkvb,
                                               __bf16* __restrict__ qkvb) {
  int lane = threadIdx.x;
  int m0 = blockIdx.x * 16;
  int o64 = blockIdx.y * 64;
  v8f acc[4] = {};
  for (int k0 = 0; k0 < Cc; k0 += 32) {
    v16bf a = load_a_f32(x, Cc, m0, k0, lane);
#pragma unroll
    for (int t = 0; t < 4; ++t) {
      v16bf bm = load_b_nt(wqkvb, Cc, o64 + t * 16, k0, lane);
      acc[t] = wmma_bf16(a, bm, acc[t]);
    }
  }
#pragma unroll
  for (int t = 0; t < 4; ++t) {
    int o = o64 + t * 16 + (lane & 15);
    int which = o / Cc;                      // uniform within a 16-wide tile
    int rest = o - which * Cc;
    int h = rest >> 6, d = rest & 63;
#pragma unroll
    for (int r = 0; r < 8; ++r) {
      int m = m0 + r + ((lane >> 4) << 3);
      int b = m / Nn, n = m - b * Nn;
      size_t head = ((size_t)which * BH + (size_t)b * Hh + h) * ((size_t)NK * Dd);
      size_t dst = (which == 2) ? head + (size_t)d * NK + n    // V: [d][n]
                                : head + (size_t)n * Dd + d;   // Q,K: [n][d]
      qkvb[dst] = (__bf16)acc[t][r];
    }
  }
}

// ---------------- K3: rel-pos score bias, i-grouped GEMM (WMMA) ----------------
// bias[bh][i][k] = sum_d q[bh][i][d] * rpk[i][k][d]

__global__ void __launch_bounds__(32) bias_gemm(const __bf16* __restrict__ qkvb,
                                                const __bf16* __restrict__ rpkb,
                                                __bf16* __restrict__ pbuf) {
  int lane = threadIdx.x;
  int i   = blockIdx.x;
  int bh0 = blockIdx.y * 16;
  int kc0 = blockIdx.z * 16;
  const __bf16* Qi = qkvb + (size_t)i * Dd;           // row stride NK*Dd over bh
  const __bf16* Ri = rpkb + (size_t)i * NK * Dd;      // [NK][Dd] row-major (NT operand)
  v8f acc = {};
  acc = wmma_bf16(load_a_bf16(Qi, NK * Dd, bh0, 0,  lane), load_b_nt(Ri, Dd, kc0, 0,  lane), acc);
  acc = wmma_bf16(load_a_bf16(Qi, NK * Dd, bh0, 32, lane), load_b_nt(Ri, Dd, kc0, 32, lane), acc);
#pragma unroll
  for (int r = 0; r < 8; ++r) {
    int bh = bh0 + r + ((lane >> 4) << 3);
    int k  = kc0 + (lane & 15);
    pbuf[((size_t)bh * NQ + i) * NK + k] = (__bf16)acc[r];
  }
}

// ---------------- K4: fused scores + softmax per (b,h) ----------------
// Whole 208x224 fp32 score matrix in LDS (CDNA5: 320KB/WGP). Q.K^T via WMMA from LDS,
// bias added from pbuf, masked softmax, probabilities written back over pbuf as bf16.

constexpr size_t SMEM_S  = (size_t)NQ * NK * 4;   // 186368
constexpr size_t SMEM_Q  = (size_t)NQ * Dd * 2;   //  26624
constexpr size_t SMEM_K  = (size_t)NK * Dd * 2;   //  28672
constexpr size_t SMEM_BYTES = SMEM_S + SMEM_Q + SMEM_K;  // 241664 < 320KB

__global__ void __launch_bounds__(256) attn_softmax(const __bf16* __restrict__ qkvb,
                                                    __bf16* __restrict__ pbuf) {
  extern __shared__ char smem[];
  float*  sS = (float*)smem;
  __bf16* sQ = (__bf16*)(smem + SMEM_S);
  __bf16* sK = (__bf16*)(smem + SMEM_S + SMEM_Q);

  int tid = threadIdx.x;
  int bh = blockIdx.x;
  const __bf16* Q = qkvb + (size_t)bh * NK * Dd;
  const __bf16* K = qkvb + ((size_t)BH + bh) * NK * Dd;
  for (int idx = tid; idx < NQ * Dd; idx += 256) sQ[idx] = Q[idx];  // rows >=197 are zero
  for (int idx = tid; idx < NK * Dd; idx += 256) sK[idx] = K[idx];
  __syncthreads();

  int wave = tid >> 5, lane = tid & 31;
  __bf16* Pb = pbuf + (size_t)bh * NQ * NK;  // bias in, probabilities out

  for (int t = wave; t < 13 * 14; t += 8) {  // wave-uniform: EXEC all-ones for WMMA
    int ti = t / 14, tj = t % 14;
    v8f acc = {};
    acc = wmma_bf16(load_a_bf16(sQ, Dd, ti * 16, 0,  lane), load_b_nt(sK, Dd, tj * 16, 0,  lane), acc);
    acc = wmma_bf16(load_a_bf16(sQ, Dd, ti * 16, 32, lane), load_b_nt(sK, Dd, tj * 16, 32, lane), acc);
#pragma unroll
    for (int r = 0; r < 8; ++r) {
      int row = ti * 16 + r + ((lane >> 4) << 3);
      int col = tj * 16 + (lane & 15);
      float bias = (float)Pb[(size_t)row * NK + col];
      float s = SCALE * (acc[r] + bias);
      if (col >= Nn) s = -1e30f;
      sS[row * NK + col] = s;
    }
  }
  __syncthreads();

  if (tid < NQ) {
    float* rowp = sS + (size_t)tid * NK;
    float mx = -1e30f;
    for (int k = 0; k < Nn; ++k) mx = fmaxf(mx, rowp[k]);
    float sum = 0.f;
    for (int k = 0; k < Nn; ++k) { float e = __expf(rowp[k] - mx); rowp[k] = e; sum += e; }
    float inv = 1.f / sum;
    for (int k = 0; k < Nn; ++k) rowp[k] *= inv;
    for (int k = Nn; k < NK; ++k) rowp[k] = 0.f;   // padded keys contribute exactly zero
  }
  __syncthreads();

  for (int idx = tid; idx < NQ * NK; idx += 256) Pb[idx] = (__bf16)sS[idx];
}

// ---------------- K5: out = P @ V, bh-grouped GEMM (WMMA) ----------------
// V is stored transposed [bh][d][n] -> B operand via contiguous NT loads.

__global__ void __launch_bounds__(32) pv_gemm(const __bf16* __restrict__ pbuf,
                                              const __bf16* __restrict__ qkvb,
                                              __bf16* __restrict__ obuf) {
  int lane = threadIdx.x;
  int bh = blockIdx.x;
  int n0 = blockIdx.y * 16;
  const __bf16* P  = pbuf + (size_t)bh * NQ * NK;
  const __bf16* Vt = qkvb + ((size_t)2 * BH + bh) * NK * Dd;  // [Dd][NK]
  v8f acc[4] = {};
  for (int k0 = 0; k0 < NK; k0 += 32) {
    v16bf a = load_a_bf16(P, NK, n0, k0, lane);
#pragma unroll
    for (int t = 0; t < 4; ++t)
      acc[t] = wmma_bf16(a, load_b_nt(Vt, NK, t * 16, k0, lane), acc[t]);
  }
  int b = bh / Hh, h = bh - b * Hh;
#pragma unroll
  for (int t = 0; t < 4; ++t) {
#pragma unroll
    for (int r = 0; r < 8; ++r) {
      int n = n0 + r + ((lane >> 4) << 3);
      if (n < Nn) {
        int d = t * 16 + (lane & 15);
        obuf[((size_t)b * Nn + n) * Cc + h * Dd + d] = (__bf16)acc[t][r];
      }
    }
  }
}

// ---------------- K6: out += P @ rpv, i-grouped GEMM (WMMA) ----------------
// rpv is stored transposed [i][d][k] -> B operand via contiguous NT loads.

__global__ void __launch_bounds__(32) rpv_gemm(const __bf16* __restrict__ pbuf,
                                               const __bf16* __restrict__ rpvt,
                                               __bf16* __restrict__ obuf) {
  int lane = threadIdx.x;
  int i   = blockIdx.x;
  int bh0 = blockIdx.y * 16;
  int d0  = blockIdx.z * 16;
  const __bf16* Pi = pbuf + (size_t)i * NK;          // row stride NQ*NK over bh
  const __bf16* Ri = rpvt + (size_t)i * Dd * NK;     // [Dd][NK]
  v8f acc = {};
  for (int k0 = 0; k0 < NK; k0 += 32)
    acc = wmma_bf16(load_a_bf16(Pi, NQ * NK, bh0, k0, lane),
                    load_b_nt(Ri, NK, d0, k0, lane), acc);
#pragma unroll
  for (int r = 0; r < 8; ++r) {
    int bh = bh0 + r + ((lane >> 4) << 3);
    int b = bh / Hh, h = bh - b * Hh;
    int d = d0 + (lane & 15);
    size_t o = ((size_t)b * Nn + i) * Cc + h * Dd + d;
    obuf[o] = (__bf16)((float)obuf[o] + acc[r]);
  }
}

// ---------------- K7: output projection GEMM (WMMA) + bias ----------------

__global__ void __launch_bounds__(32) proj_gemm(const __bf16* __restrict__ obuf,
                                                const __bf16* __restrict__ wprojb,
                                                const float* __restrict__ bproj,
                                                float* __restrict__ out) {
  int lane = threadIdx.x;
  int m0 = blockIdx.x * 16;
  int o0 = blockIdx.y * 64;
  v8f acc[4] = {};
  for (int k0 = 0; k0 < Cc; k0 += 32) {
    v16bf a = load_a_bf16(obuf, Cc, m0, k0, lane);
#pragma unroll
    for (int t = 0; t < 4; ++t)
      acc[t] = wmma_bf16(a, load_b_nt(wprojb, Cc, o0 + t * 16, k0, lane), acc[t]);
  }
#pragma unroll
  for (int t = 0; t < 4; ++t) {
    int o = o0 + t * 16 + (lane & 15);
    float bias = bproj[o];
#pragma unroll
    for (int r = 0; r < 8; ++r) {
      int m = m0 + r + ((lane >> 4) << 3);
      out[(size_t)m * Cc + o] = acc[t][r] + bias;
    }
  }
}

// ---------------- host launch ----------------

extern "C" void kernel_launch(void* const* d_in, const int* in_sizes, int n_in,
                              void* d_out, int out_size, void* d_ws, size_t ws_size,
                              hipStream_t stream) {
  const float* x     = (const float*)d_in[0];
  const float* wqkv  = (const float*)d_in[1];
  const float* wproj = (const float*)d_in[2];
  const float* bproj = (const float*)d_in[3];
  const float* rkv   = (const float*)d_in[4];
  const float* rkh   = (const float*)d_in[5];
  const float* rvv   = (const float*)d_in[6];
  const float* rvh   = (const float*)d_in[7];
  float* out = (float*)d_out;

  auto al = [](size_t v) { return (v + 255) & ~(size_t)255; };
  char* ws = (char*)d_ws;
  size_t off = 0;
  __bf16* wqkvb  = (__bf16*)(ws + off); off = al(off + (size_t)O3 * Cc * 2);
  __bf16* wprojb = (__bf16*)(ws + off); off = al(off + (size_t)Cc * Cc * 2);
  __bf16* qkvb   = (__bf16*)(ws + off); off = al(off + (size_t)3 * BH * NK * Dd * 2);
  __bf16* rpkb   = (__bf16*)(ws + off); off = al(off + (size_t)Nn * NK * Dd * 2);
  __bf16* rpvt   = (__bf16*)(ws + off); off = al(off + (size_t)Nn * NK * Dd * 2);
  __bf16* pbuf   = (__bf16*)(ws + off); off = al(off + (size_t)BH * NQ * NK * 2);  // bias, then P
  __bf16* obuf   = (__bf16*)(ws + off); off = al(off + (size_t)Mm * Cc * 2);

  // allow 242KB dynamic LDS for the fused attention kernel
  hipFuncSetAttribute((const void*)attn_softmax,
                      hipFuncAttributeMaxDynamicSharedMemorySize, (int)SMEM_BYTES);

  prep_kernel<<<2048, 256, 0, stream>>>(wqkv, wproj, wqkvb, wprojb, qkvb);
  relemb_kernel<<<(Nn * NK + 255) / 256, 256, 0, stream>>>(rkv, rkh, rvv, rvh, rpkb, rpvt);
  qkv_gemm<<<dim3(Mm / 16, O3 / 64), 32, 0, stream>>>(x, wqkvb, qkvb);
  bias_gemm<<<dim3(Nn, BH / 16, NK / 16), 32, 0, stream>>>(qkvb, rpkb, pbuf);
  attn_softmax<<<BH, 256, SMEM_BYTES, stream>>>(qkvb, pbuf);
  pv_gemm<<<dim3(BH, NQ / 16), 32, 0, stream>>>(pbuf, qkvb, obuf);
  rpv_gemm<<<dim3(Nn, BH / 16, Dd / 16), 32, 0, stream>>>(pbuf, rpvt, obuf);
  proj_gemm<<<dim3(Mm / 16, Cc / 64), 32, 0, stream>>>(obuf, wprojb, bproj, out);
}
